// HeteroGNN_79663053406767
// MI455X (gfx1250) — compile-verified, matrix operand order checked by reading, not run on hardware
//
#include <hip/hip_runtime.h>

#define HID 128
#define NPP 50000
#define NEE 75000
#define NFF 100000
#define EE  200000
#define NG  16
#define LAYERS 6

typedef __attribute__((ext_vector_type(16))) __bf16 v16bf;
typedef __attribute__((ext_vector_type(8)))  float  v8f;
typedef __attribute__((ext_vector_type(4)))  unsigned int v4u;
typedef __attribute__((ext_vector_type(8)))  int v8i;
typedef __attribute__((ext_vector_type(4)))  int v4i;

// ---------------- zero ----------------
__global__ void zero_kernel(float* __restrict__ p, size_t n) {
  size_t t = (size_t)blockIdx.x * blockDim.x + threadIdx.x;
  if (t < n) p[t] = 0.f;
}

// ------------- weight pack: fp32 (sum of up to 3) -> bf16 in WMMA B layout -------------
// out[((kt*8+nt)*32+lane)*16+i] = W[K][N], K = kt*32 + (lane&16?8:0) + (i<8?i:i+8),
// N = nt*16 + (lane&15)   (ISA 7.12.2 16-bit B operand striping, wave32)
__global__ __launch_bounds__(256) void pack_weight_kernel(
    const float* __restrict__ W0, const float* __restrict__ W1,
    const float* __restrict__ W2, __bf16* __restrict__ out) {
  int o = blockIdx.x * 256 + threadIdx.x;        // 0..16383
  int i = o & 15, lane = (o >> 4) & 31, nt = (o >> 9) & 7, kt = o >> 12;
  int K = kt * 32 + ((lane & 16) ? 8 : 0) + (i < 8 ? i : i + 8);
  int N = nt * 16 + (lane & 15);
  float w = W0[K * HID + N];
  if (W1) w += W1[K * HID + N];
  if (W2) w += W2[K * HID + N];
  out[o] = (__bf16)w;
}

__global__ void combine_bias_kernel(const float* __restrict__ b0,
                                    const float* __restrict__ b1,
                                    const float* __restrict__ b2,
                                    float* __restrict__ out) {
  int j = threadIdx.x;
  float v = b0[j];
  if (b1) v += b1[j];
  if (b2) v += b2[j];
  out[j] = v;
}

// ------------- encoders: x[din] -> relu(x@W1+b1) -> @W2+b2 (tiny din, VALU) -------------
__global__ __launch_bounds__(128) void encoder_kernel(
    const float* __restrict__ x, int din,
    const float* __restrict__ W1, const float* __restrict__ b1,
    const float* __restrict__ W2, const float* __restrict__ b2,
    float* __restrict__ h, int n) {
  __shared__ float xs[8];
  __shared__ float h1[HID];
  int row = blockIdx.x, j = threadIdx.x;
  if (row >= n) return;
  if (j < din) xs[j] = x[(size_t)row * din + j];
  __syncthreads();
  float a = b1[j];
  for (int k = 0; k < din; ++k) a += xs[k] * W1[k * HID + j];
  h1[j] = fmaxf(a, 0.f);
  __syncthreads();
  float o = b2[j];
  #pragma unroll 8
  for (int k = 0; k < HID; ++k) o += h1[k] * W2[k * HID + j];
  h[(size_t)row * HID + j] = o;
}

// ------------- scatter-mean numerator + counts (L2-resident atomics) -------------
__global__ __launch_bounds__(256) void scatter_add_kernel(
    const float* __restrict__ hsrc, const int* __restrict__ src,
    const int* __restrict__ dst, float* __restrict__ agg,
    float* __restrict__ cnt, int ne) {
  int t = blockIdx.x * 256 + threadIdx.x;
  int e = t >> 5;
  if (e >= ne) return;
  int lane = t & 31;
  int s = src[e], d = dst[e];
  const float4 v = *(const float4*)(hsrc + (size_t)s * HID + lane * 4);
  float* p = agg + (size_t)d * HID + lane * 4;
  atomicAdd(p + 0, v.x); atomicAdd(p + 1, v.y);
  atomicAdd(p + 2, v.z); atomicAdd(p + 3, v.w);
  if (lane == 0) atomicAdd(cnt + d, 1.f);
}

// ------------- WMMA GEMM: C[n,128] (=|+=) (A * scale) @ Wpacked (+bias) -------------
// Weights (32KB, bf16, WMMA B layout) staged once per workgroup into LDS via the
// Tensor Data Mover (tensor_load_to_lds + s_wait_tensorcnt). Per K-step all 8 B
// tiles are batch-loaded from LDS first, then 8 back-to-back WMMAs consume them,
// so ds_load latency hides under matrix work. One wave = 16x128 tile (32 WMMAs).
__global__ __launch_bounds__(256) void sage_gemm_wmma(
    const float* __restrict__ A, const float* __restrict__ cntv,
    const __bf16* __restrict__ Wp, const float* __restrict__ bias,
    float* __restrict__ C, int n, int accumulate) {
  __shared__ __bf16 sW[HID * HID];            // 32 KB staged weight tile

#if defined(__gfx1250__) && __has_builtin(__builtin_amdgcn_tensor_load_to_lds)
  if (threadIdx.x < 32) {                     // wave 0 drives the TDM
    unsigned lds_off = (unsigned)(unsigned long long)(void*)&sW[0];
    unsigned long long ga = (unsigned long long)Wp;
    // D# group0: count=1 (valid), lds_addr, global_addr[56:0], type=2 (image)
    v4u g0;
    g0.x = 1u;
    g0.y = lds_off;
    g0.z = (unsigned)ga;
    g0.w = (unsigned)((ga >> 32) & 0x01FFFFFFull) | (2u << 30);
    // D# group1: wg_mask=0, data_size=1 (2B), 1-row tile of 16384 elements
    v8i g1;
    g1[0] = (int)(1u << 16);                        // data_size=2B
    g1[1] = (int)((16384u & 0xFFFFu) << 16);        // tensor_dim0[15:0]
    g1[2] = (int)((16384u >> 16) | (1u << 16));     // tensor_dim0[31:16] | tensor_dim1=1
    g1[3] = (int)(16384u << 16);                    // tile_dim0 = 16384
    g1[4] = (int)(1u);                              // tile_dim1 = 1, tile_dim2 = 0
    g1[5] = (int)(16384u);                          // tensor_dim0_stride[31:0]
    g1[6] = 0;
    g1[7] = 0;
    v4i z4 = {0, 0, 0, 0};
    v8i z8 = {0, 0, 0, 0, 0, 0, 0, 0};
    __builtin_amdgcn_tensor_load_to_lds(g0, g1, z4, z4, z8, 0);
    __builtin_amdgcn_s_wait_tensorcnt(0);
  }
  __syncthreads();
#else
  for (int i = threadIdx.x; i < (HID * HID * 2) / 16; i += 256)
    ((uint4*)sW)[i] = ((const uint4*)Wp)[i];
  __syncthreads();
#endif

  int wave = threadIdx.x >> 5;
  int lane = threadIdx.x & 31;
  int m0 = (blockIdx.x * 8 + wave) * 16;
  if (m0 >= n) return;                       // wave-uniform: EXEC stays full for WMMA

  int col  = lane & 15;
  int half = lane >> 4;                      // K-half select for A/B striping
  int m    = m0 + col;
  bool mv  = (m < n);

  float s = 1.0f;
  if (cntv && mv) s = 1.0f / fmaxf(cntv[m], 1.0f);

  v8f acc[8];
  if (accumulate) {
    #pragma unroll
    for (int nt = 0; nt < 8; ++nt)
      #pragma unroll
      for (int v = 0; v < 8; ++v) {
        int mr = m0 + half * 8 + v;
        acc[nt][v] = (mr < n) ? C[(size_t)mr * HID + nt * 16 + col] : 0.f;
      }
  } else {
    #pragma unroll
    for (int nt = 0; nt < 8; ++nt) {
      float bv = bias ? bias[nt * 16 + col] : 0.f;
      #pragma unroll
      for (int v = 0; v < 8; ++v) acc[nt][v] = bv;
    }
  }

  const float* ap = A + (size_t)m * HID;
  if (mv) __builtin_prefetch(ap, 0, 3);      // global_prefetch of this wave's A rows

  #pragma unroll
  for (int kt = 0; kt < 4; ++kt) {
    int kb = kt * 32 + half * 8;
    v16bf a;
    if (mv) {
      float4 f0 = *(const float4*)(ap + kb);
      float4 f1 = *(const float4*)(ap + kb + 4);
      float4 f2 = *(const float4*)(ap + kb + 16);
      float4 f3 = *(const float4*)(ap + kb + 20);
      a[0]=(__bf16)(f0.x*s); a[1]=(__bf16)(f0.y*s); a[2]=(__bf16)(f0.z*s); a[3]=(__bf16)(f0.w*s);
      a[4]=(__bf16)(f1.x*s); a[5]=(__bf16)(f1.y*s); a[6]=(__bf16)(f1.z*s); a[7]=(__bf16)(f1.w*s);
      a[8]=(__bf16)(f2.x*s); a[9]=(__bf16)(f2.y*s); a[10]=(__bf16)(f2.z*s); a[11]=(__bf16)(f2.w*s);
      a[12]=(__bf16)(f3.x*s); a[13]=(__bf16)(f3.y*s); a[14]=(__bf16)(f3.z*s); a[15]=(__bf16)(f3.w*s);
    } else {
      #pragma unroll
      for (int i = 0; i < 16; ++i) a[i] = (__bf16)0.f;
    }
    // batch-load all 8 B tiles for this K-step, then issue 8 WMMAs back-to-back
    v16bf bt[8];
    #pragma unroll
    for (int nt = 0; nt < 8; ++nt)
      bt[nt] = *(const v16bf*)(sW + (((kt * 8 + nt) * 32 + lane) << 4));
    #pragma unroll
    for (int nt = 0; nt < 8; ++nt)
      acc[nt] = __builtin_amdgcn_wmma_f32_16x16x32_bf16(
          false, a, false, bt[nt], (short)0, acc[nt], false, false);
  }

  #pragma unroll
  for (int nt = 0; nt < 8; ++nt)
    #pragma unroll
    for (int v = 0; v < 8; ++v) {
      int mr = m0 + half * 8 + v;
      if (mr < n) C[(size_t)mr * HID + nt * 16 + col] = acc[nt][v];
    }
}

// ------------- h = layernorm(h + relu(o)) : one wave per row, shfl_xor reduce -------------
__global__ __launch_bounds__(256) void post_ln_kernel(
    float* __restrict__ h, const float* __restrict__ o,
    const float* __restrict__ gamma, const float* __restrict__ beta, int n) {
  int wave = threadIdx.x >> 5, lane = threadIdx.x & 31;
  int row = blockIdx.x * 8 + wave;
  if (row >= n) return;
  const float4 hv = *(const float4*)(h + (size_t)row * HID + lane * 4);
  const float4 ov = *(const float4*)(o + (size_t)row * HID + lane * 4);
  float x[4] = { hv.x + fmaxf(ov.x, 0.f), hv.y + fmaxf(ov.y, 0.f),
                 hv.z + fmaxf(ov.z, 0.f), hv.w + fmaxf(ov.w, 0.f) };
  float sum = x[0] + x[1] + x[2] + x[3];
  #pragma unroll
  for (int off = 16; off >= 1; off >>= 1) sum += __shfl_xor(sum, off, 32);
  float mu = sum * (1.0f / HID);
  float d[4], v = 0.f;
  #pragma unroll
  for (int i = 0; i < 4; ++i) { d[i] = x[i] - mu; v += d[i] * d[i]; }
  #pragma unroll
  for (int off = 16; off >= 1; off >>= 1) v += __shfl_xor(v, off, 32);
  float inv = rsqrtf(v * (1.0f / HID) + 1e-5f);
  float4 r;
  r.x = d[0] * inv * gamma[lane * 4 + 0] + beta[lane * 4 + 0];
  r.y = d[1] * inv * gamma[lane * 4 + 1] + beta[lane * 4 + 1];
  r.z = d[2] * inv * gamma[lane * 4 + 2] + beta[lane * 4 + 2];
  r.w = d[3] * inv * gamma[lane * 4 + 3] + beta[lane * 4 + 3];
  *(float4*)(h + (size_t)row * HID + lane * 4) = r;
}

// ------------- mean pool into fused[g, col_off .. col_off+127] -------------
__global__ __launch_bounds__(256) void pool_kernel(
    const float* __restrict__ h, const int* __restrict__ batch,
    float* __restrict__ fused, float* __restrict__ cnt, int n, int col_off) {
  int t = blockIdx.x * 256 + threadIdx.x;
  int node = t >> 5;
  if (node >= n) return;
  int lane = t & 31;
  int g = batch[node];
  const float4 v = *(const float4*)(h + (size_t)node * HID + lane * 4);
  float* p = fused + (size_t)g * 448 + col_off + lane * 4;
  atomicAdd(p + 0, v.x); atomicAdd(p + 1, v.y);
  atomicAdd(p + 2, v.z); atomicAdd(p + 3, v.w);
  if (lane == 0) atomicAdd(cnt + g, 1.f);
}

__global__ void pool_finalize(float* __restrict__ fused, const float* __restrict__ pcnt) {
  int t = blockIdx.x * blockDim.x + threadIdx.x;   // NG*384
  if (t >= NG * 384) return;
  int g = t / 384, j = t % 384;
  float c = pcnt[(j >> 7) * NG + g];               // 0:p 1:e 2:f
  fused[(size_t)g * 448 + j] /= fmaxf(c, 1.f);
}

// ------------- global MLP 14->64->64->64 -> fused[:,384:448] -------------
__global__ __launch_bounds__(64) void global_mlp_kernel(
    const float* __restrict__ gf,
    const float* __restrict__ W1, const float* __restrict__ b1,
    const float* __restrict__ W2, const float* __restrict__ b2,
    const float* __restrict__ W3, const float* __restrict__ b3,
    float* __restrict__ fused) {
  __shared__ float t0[14]; __shared__ float t1[64]; __shared__ float t2[64];
  int j = threadIdx.x;
  for (int row = 0; row < NG; ++row) {
    if (j < 14) t0[j] = gf[row * 14 + j];
    __syncthreads();
    float a = b1[j];
    for (int k = 0; k < 14; ++k) a += t0[k] * W1[k * 64 + j];
    t1[j] = fmaxf(a, 0.f);
    __syncthreads();
    float c = b2[j];
    for (int k = 0; k < 64; ++k) c += t1[k] * W2[k * 64 + j];
    t2[j] = fmaxf(c, 0.f);
    __syncthreads();
    float d = b3[j];
    for (int k = 0; k < 64; ++k) d += t2[k] * W3[k * 64 + j];
    fused[(size_t)row * 448 + 384 + j] = d;         // last layer: no relu
    __syncthreads();
  }
}

// ------------- decoder 448->256->256->1 -------------
__global__ __launch_bounds__(256) void decoder_kernel(
    const float* __restrict__ fused,
    const float* __restrict__ W1, const float* __restrict__ b1,
    const float* __restrict__ W2, const float* __restrict__ b2,
    const float* __restrict__ W3, const float* __restrict__ b3,
    float* __restrict__ out) {
  __shared__ float f[448]; __shared__ float h1[256];
  __shared__ float h2[256]; __shared__ float red[256];
  int j = threadIdx.x;
  for (int row = 0; row < NG; ++row) {
    for (int k = j; k < 448; k += 256) f[k] = fused[(size_t)row * 448 + k];
    __syncthreads();
    float a = b1[j];
    for (int k = 0; k < 448; ++k) a += f[k] * W1[k * 256 + j];
    h1[j] = fmaxf(a, 0.f);
    __syncthreads();
    float c = b2[j];
    for (int k = 0; k < 256; ++k) c += h1[k] * W2[k * 256 + j];
    h2[j] = fmaxf(c, 0.f);
    __syncthreads();
    red[j] = h2[j] * W3[j];
    __syncthreads();
    for (int s2 = 128; s2 > 0; s2 >>= 1) {
      if (j < s2) red[j] += red[j + s2];
      __syncthreads();
    }
    if (j == 0) out[row] = red[0] + b3[0];
    __syncthreads();
  }
}

// =====================================================================
extern "C" void kernel_launch(void* const* d_in, const int* in_sizes, int n_in,
                              void* d_out, int out_size, void* d_ws, size_t ws_size,
                              hipStream_t stream) {
  (void)in_sizes; (void)n_in; (void)out_size; (void)ws_size;
  auto F = [&](int i) { return (const float*)d_in[i]; };
  auto I = [&](int i) { return (const int*)d_in[i]; };

  // top-level inputs (insertion order)
  const float* x_point = F(0); const float* x_edge = F(1); const float* x_face = F(2);
  const float* gfeat = F(3);
  const int* batch_p = I(4); const int* batch_e = I(5); const int* batch_f = I(6);
  // edge lists 7..22: pp(7,8) fp(9,10) ep(11,12) pf(13,14) ef(15,16) ff(17,18) pe(19,20) fe(21,22)

  // params as sorted JAX tree leaves from index 23:
  // decoder 23..28, enc_edge 29..32, enc_face 33..36, enc_point 37..40, global 41..46,
  // layers L: base 47+30L: ln_e(g,b) ln_f(g,b) ln_p(g,b), rels sorted
  // ef(+6..8) ep(+9..11) fe(+12..14) ff(+15..17) fp(+18..20) pe(+21..23) pf(+24..26) pp(+27..29)

  char* wsb = (char*)d_ws;
  size_t off = 0;
  auto allocf = [&](size_t nelem) -> float* {
    off = (off + 255) & ~(size_t)255;
    float* p = (float*)(wsb + off);
    off += nelem * sizeof(float);
    return p;
  };
  float* hp   = allocf((size_t)NPP * HID);
  float* he   = allocf((size_t)NEE * HID);
  float* hf   = allocf((size_t)NFF * HID);
  float* op   = allocf((size_t)NPP * HID);
  float* of_  = allocf((size_t)NFF * HID);
  float* oe   = allocf((size_t)NEE * HID);
  float* agg  = allocf((size_t)NFF * HID);
  float* cnt  = allocf((size_t)NFF);
  float* fused= allocf((size_t)NG * 448);
  float* pcnt = allocf((size_t)3 * NG);
  float* pbias= allocf((size_t)LAYERS * 3 * HID);
  off = (off + 255) & ~(size_t)255;
  __bf16* pw = (__bf16*)(wsb + off);
  off += (size_t)LAYERS * 11 * 16384 * sizeof(__bf16);

  // ---- pack weights (bf16, WMMA B layout) + combined biases ----
  for (int L = 0; L < LAYERS; ++L) {
    int lb = 47 + 30 * L;
    const float *Wl_ef=F(lb+6),  *bl_ef=F(lb+7),  *Wr_ef=F(lb+8);
    const float *Wl_ep=F(lb+9),  *bl_ep=F(lb+10), *Wr_ep=F(lb+11);
    const float *Wl_fe=F(lb+12), *bl_fe=F(lb+13), *Wr_fe=F(lb+14);
    const float *Wl_ff=F(lb+15), *bl_ff=F(lb+16), *Wr_ff=F(lb+17);
    const float *Wl_fp=F(lb+18), *bl_fp=F(lb+19), *Wr_fp=F(lb+20);
    const float *Wl_pe=F(lb+21), *bl_pe=F(lb+22), *Wr_pe=F(lb+23);
    const float *Wl_pf=F(lb+24), *bl_pf=F(lb+25), *Wr_pf=F(lb+26);
    const float *Wl_pp=F(lb+27), *bl_pp=F(lb+28), *Wr_pp=F(lb+29);
    __bf16* base = pw + (size_t)L * 11 * 16384;
    // combined self-transforms per dst type (sum of Wr over rels sharing dst)
    pack_weight_kernel<<<64,256,0,stream>>>(Wr_pp, Wr_fp, Wr_ep, base + 0*16384);  // dst p
    pack_weight_kernel<<<64,256,0,stream>>>(Wr_pf, Wr_ef, Wr_ff, base + 1*16384);  // dst f
    pack_weight_kernel<<<64,256,0,stream>>>(Wr_pe, Wr_fe, nullptr, base + 2*16384);// dst e
    pack_weight_kernel<<<64,256,0,stream>>>(Wl_pp, nullptr, nullptr, base + 3*16384);
    pack_weight_kernel<<<64,256,0,stream>>>(Wl_fp, nullptr, nullptr, base + 4*16384);
    pack_weight_kernel<<<64,256,0,stream>>>(Wl_ep, nullptr, nullptr, base + 5*16384);
    pack_weight_kernel<<<64,256,0,stream>>>(Wl_pf, nullptr, nullptr, base + 6*16384);
    pack_weight_kernel<<<64,256,0,stream>>>(Wl_ef, nullptr, nullptr, base + 7*16384);
    pack_weight_kernel<<<64,256,0,stream>>>(Wl_ff, nullptr, nullptr, base + 8*16384);
    pack_weight_kernel<<<64,256,0,stream>>>(Wl_pe, nullptr, nullptr, base + 9*16384);
    pack_weight_kernel<<<64,256,0,stream>>>(Wl_fe, nullptr, nullptr, base + 10*16384);
    combine_bias_kernel<<<1,128,0,stream>>>(bl_pp, bl_fp, bl_ep, pbias + (L*3+0)*HID);
    combine_bias_kernel<<<1,128,0,stream>>>(bl_pf, bl_ef, bl_ff, pbias + (L*3+1)*HID);
    combine_bias_kernel<<<1,128,0,stream>>>(bl_pe, bl_fe, nullptr, pbias + (L*3+2)*HID);
  }

  // ---- encoders ----
  encoder_kernel<<<NPP,128,0,stream>>>(x_point, 6, F(37), F(38), F(39), F(40), hp, NPP);
  encoder_kernel<<<NEE,128,0,stream>>>(x_edge,  2, F(29), F(30), F(31), F(32), he, NEE);
  encoder_kernel<<<NFF,128,0,stream>>>(x_face,  7, F(33), F(34), F(35), F(36), hf, NFF);

  auto gemm = [&](const float* A, const float* cv, const __bf16* W, const float* b,
                  float* C, int n, int accum) {
    sage_gemm_wmma<<<(n + 127) / 128, 256, 0, stream>>>(A, cv, W, b, C, n, accum);
  };
  auto scatter = [&](const float* hsrc, int si, int di, int ndst) {
    zero_kernel<<<(int)(((size_t)ndst * HID + 255) / 256), 256, 0, stream>>>(agg, (size_t)ndst * HID);
    zero_kernel<<<(ndst + 255) / 256, 256, 0, stream>>>(cnt, (size_t)ndst);
    scatter_add_kernel<<<(EE * 32 + 255) / 256, 256, 0, stream>>>(hsrc, I(si), I(di), agg, cnt, EE);
  };

  // ---- 6 message-passing layers ----
  for (int L = 0; L < LAYERS; ++L) {
    __bf16* base = pw + (size_t)L * 11 * 16384;
    // dst = point
    gemm(hp, nullptr, base + 0*16384, pbias + (L*3+0)*HID, op, NPP, 0);
    scatter(hp, 7,  8,  NPP); gemm(agg, cnt, base + 3*16384, nullptr, op, NPP, 1);  // pp
    scatter(hf, 9,  10, NPP); gemm(agg, cnt, base + 4*16384, nullptr, op, NPP, 1);  // fp
    scatter(he, 11, 12, NPP); gemm(agg, cnt, base + 5*16384, nullptr, op, NPP, 1);  // ep
    // dst = face
    gemm(hf, nullptr, base + 1*16384, pbias + (L*3+1)*HID, of_, NFF, 0);
    scatter(hp, 13, 14, NFF); gemm(agg, cnt, base + 6*16384, nullptr, of_, NFF, 1); // pf
    scatter(he, 15, 16, NFF); gemm(agg, cnt, base + 7*16384, nullptr, of_, NFF, 1); // ef
    scatter(hf, 17, 18, NFF); gemm(agg, cnt, base + 8*16384, nullptr, of_, NFF, 1); // ff
    // dst = edge
    gemm(he, nullptr, base + 2*16384, pbias + (L*3+2)*HID, oe, NEE, 0);
    scatter(hp, 19, 20, NEE); gemm(agg, cnt, base + 9*16384,  nullptr, oe, NEE, 1); // pe
    scatter(hf, 21, 22, NEE); gemm(agg, cnt, base + 10*16384, nullptr, oe, NEE, 1); // fe
    // residual + relu + layernorm (after all o's computed from old h's)
    int lb = 47 + 30 * L;
    post_ln_kernel<<<(NPP + 7) / 8, 256, 0, stream>>>(hp, op,  F(lb+4), F(lb+5), NPP);
    post_ln_kernel<<<(NFF + 7) / 8, 256, 0, stream>>>(hf, of_, F(lb+2), F(lb+3), NFF);
    post_ln_kernel<<<(NEE + 7) / 8, 256, 0, stream>>>(he, oe,  F(lb+0), F(lb+1), NEE);
  }

  // ---- pooling + heads ----
  zero_kernel<<<(NG * 448 + 255) / 256, 256, 0, stream>>>(fused, (size_t)NG * 448);
  zero_kernel<<<1, 256, 0, stream>>>(pcnt, (size_t)3 * NG);
  pool_kernel<<<(NPP * 32 + 255) / 256, 256, 0, stream>>>(hp, batch_p, fused, pcnt + 0,      NPP, 0);
  pool_kernel<<<(NEE * 32 + 255) / 256, 256, 0, stream>>>(he, batch_e, fused, pcnt + NG,     NEE, 128);
  pool_kernel<<<(NFF * 32 + 255) / 256, 256, 0, stream>>>(hf, batch_f, fused, pcnt + 2 * NG, NFF, 256);
  pool_finalize<<<(NG * 384 + 255) / 256, 256, 0, stream>>>(fused, pcnt);
  global_mlp_kernel<<<1, 64, 0, stream>>>(gfeat, F(41), F(42), F(43), F(44), F(45), F(46), fused);
  decoder_kernel<<<1, 256, 0, stream>>>(fused, F(23), F(24), F(25), F(26), F(27), F(28),
                                        (float*)d_out);
}
